// EncoderImagePrecomp_80908593922206
// MI455X (gfx1250) — compile-verified, matrix operand order checked by reading, not run on hardware
//
#include <hip/hip_runtime.h>
#include <hip/hip_bf16.h>

// ---------------------------------------------------------------------------
// Problem constants (from the reference)
// ---------------------------------------------------------------------------
#define BR       9216      // B*R rows
#define IMG_DIM  2048
#define TAG_DIM  768
#define DIM1     2816      // IMG_DIM + TAG_DIM  (K of GEMM 1)
#define PE       200       // POS_EMBED          (N of GEMM 1)
#define NPAD1    208       // 13 tiles of 16
#define FC_IN    3016      // IMG + PE + TAG     (K of GEMM 2)
#define FC_KPAD  3040      // padded to mult of 32
#define ES       1024      // EMBED_SIZE         (N of GEMM 2)
#define BOX      15

#define LDSK     40        // LDS row stride in bf16 elems (64B data + 16B pad -> 80B rows)

typedef __attribute__((ext_vector_type(16))) __bf16        v16bf;
typedef __attribute__((ext_vector_type(8)))  float         v8f;
typedef __attribute__((ext_vector_type(4))) unsigned int   u32x4;
typedef __attribute__((ext_vector_type(8)))  int           i32x8;
typedef __attribute__((ext_vector_type(4)))  int           i32x4;

// ---------------------------------------------------------------------------
// f32 -> (hi, lo) bf16 split with RNE rounding (bf16x3 scheme)
// ---------------------------------------------------------------------------
__device__ __forceinline__ void split_store(unsigned short* hi, unsigned short* lo, float x) {
    union { float f; unsigned int u; } a; a.f = x;
    unsigned int r = (a.u + 0x7FFFu + ((a.u >> 16) & 1u)) & 0xFFFF0000u;
    *hi = (unsigned short)(r >> 16);
    union { unsigned int u; float f; } b; b.u = r;
    float res = x - b.f;
    union { float f; unsigned int u; } c; c.f = res;
    unsigned int r2 = (c.u + 0x7FFFu + ((c.u >> 16) & 1u)) & 0xFFFF0000u;
    *lo = (unsigned short)(r2 >> 16);
}

__device__ __forceinline__ void split_store4(unsigned short* hiB, unsigned short* loB,
                                             int off, float4 x) {
    split_store(hiB + off + 0, loB + off + 0, x.x);
    split_store(hiB + off + 1, loB + off + 1, x.y);
    split_store(hiB + off + 2, loB + off + 2, x.z);
    split_store(hiB + off + 3, loB + off + 3, x.w);
}

__device__ __forceinline__ float4 ld4(const float* p) {
    return *reinterpret_cast<const float4*>(p);
}

// ---------------------------------------------------------------------------
// WMMA fragment load from an LDS tile stored as [row][LDSK] (bf16).
// CDNA5 16-bit operand layout: lanes 0-15: row=lane, K 0-7 / 16-23;
// lanes 16-31: row=lane-16, K 8-15 / 24-31.  Two ds_load_b128 per fragment.
// ---------------------------------------------------------------------------
__device__ __forceinline__ v16bf load_frag(const unsigned short* base, int lane) {
    const unsigned short* p = base + (lane & 15) * LDSK + ((lane & 16) ? 8 : 0);
    union { u32x4 q[2]; v16bf v; } u;
    u.q[0] = *reinterpret_cast<const u32x4*>(p);
    u.q[1] = *reinterpret_cast<const u32x4*>(p + 16);
    return u.v;
}

#define WMMA_BF16(A, B, C) \
    __builtin_amdgcn_wmma_f32_16x16x32_bf16(false, (A), false, (B), (short)0, (C), false, false)

// ---------------------------------------------------------------------------
// TDM: DMA a [rows][32] bf16 tile (row stride kStride elems in global) into
// LDS at ldsOff, writing rows at 80B pitch (64B data + 16B pad) == LDSK.
// D# encoding per CDNA5 ISA ch.8: data_size=2B, pad_interval=16DW(64B),
// pad_amount=4DW(16B), tile_dim0=32, tile_dim1=rows, tensor dims large
// (tiles are always fully in-bounds in the pre-padded weight planes).
// This toolchain exposes the 6-arg builtin; the 5th (extra) group is
// zero-filled, cpol = 0.
// ---------------------------------------------------------------------------
__device__ __forceinline__ void tdm_load_tile(unsigned ldsOff, const unsigned short* gsrc,
                                              int rows, int kStride) {
    const unsigned long long ga = (unsigned long long)gsrc;
    u32x4 g0;
    g0[0] = 1u;                                             // count=1, user mode
    g0[1] = ldsOff;                                         // lds_addr
    g0[2] = (unsigned)(ga & 0xFFFFFFFFull);                 // global_addr[31:0]
    g0[3] = ((unsigned)(ga >> 32) & 0x01FFFFFFu) | 0x80000000u; // addr[56:32] | type=2
    i32x8 g1;
    g1[0] = (1 << 16) | (1 << 20) | (3 << 22) | (3 << 25);  // data_size=2B, pad en, 64B/16B
    g1[1] = 0;                                              // tensor_dim0 = 1<<20 (low16=0)
    g1[2] = (int)((1u << 20) >> 16);                        // tdim0 hi | tdim1 low16(=0)
    g1[3] = (int)(((1u << 20) >> 16) | (32u << 16));        // tdim1 hi | tile_dim0=32
    g1[4] = rows;                                           // tile_dim1 | tile_dim2=0
    g1[5] = kStride;                                        // tensor_dim0_stride (elems)
    g1[6] = 0;
    g1[7] = 0;
    i32x4 g2; g2[0] = 0x7FFFFFFF; g2[1] = 0x7FFFFFFF; g2[2] = 0; g2[3] = 0;
    i32x4 g3; g3[0] = 0;          g3[1] = 0x7FFF0000; g3[2] = 0x7FFF; g3[3] = 0;
    i32x8 g4; g4[0] = 0; g4[1] = 0; g4[2] = 0; g4[3] = 0;
              g4[4] = 0; g4[5] = 0; g4[6] = 0; g4[7] = 0;
    __builtin_amdgcn_tensor_load_to_lds(g0, g1, g2, g3, g4, 0);
}

// ===========================================================================
// Weight pre-split kernels: f32 -> persistent bf16 hi/lo planes in ws
// (run once per launch; weights are L2-resident, re-used by all GEMM blocks)
// ===========================================================================
__global__ __launch_bounds__(256) void k_split_fcW(const float* __restrict__ fc_W,
                                                   unsigned short* __restrict__ whi,
                                                   unsigned short* __restrict__ wlo) {
    const int total = ES * FC_KPAD;
    for (int i = blockIdx.x * 256 + threadIdx.x; i < total; i += gridDim.x * 256) {
        const int n = i / FC_KPAD;
        const int k = i - n * FC_KPAD;
        const float v = (k < FC_IN) ? fc_W[(size_t)n * FC_IN + k] : 0.f;
        split_store(whi + i, wlo + i, v);
    }
}

__global__ __launch_bounds__(256) void k_split_attnW(const float* __restrict__ attn_W,
                                                     unsigned short* __restrict__ whi,
                                                     unsigned short* __restrict__ wlo) {
    const int total = NPAD1 * DIM1;
    for (int i = blockIdx.x * 256 + threadIdx.x; i < total; i += gridDim.x * 256) {
        const int n = i / DIM1;
        const int k = i - n * DIM1;
        const float v = (n < PE) ? attn_W[(size_t)n * DIM1 + k] : 0.f;
        split_store(whi + i, wlo + i, v);
    }
}

// ===========================================================================
// Kernel 1: region = X1 @ attn_W^T      (9216 x 2816) x (2816 x 200)
//   BM=128 (8 waves x 16), BN=208 (13 tiles), BK=32; B tiles via TDM
// ===========================================================================
__global__ __launch_bounds__(256) void k_region(const float* __restrict__ images,
                                                const float* __restrict__ tags,
                                                const unsigned short* __restrict__ whi,
                                                const unsigned short* __restrict__ wlo,
                                                float* __restrict__ region) {
    __shared__ unsigned short sA[2][128 * LDSK];   // [hi/lo][m][k]
    __shared__ unsigned short sB[2][NPAD1 * LDSK]; // [hi/lo][n][k]

    const int tid  = threadIdx.x;
    const int wave = tid >> 5;
    const int lane = tid & 31;
    const int mBase = blockIdx.x * 128;

    const unsigned ldsB0 = (unsigned)(size_t)&sB[0][0];
    const unsigned ldsB1 = (unsigned)(size_t)&sB[1][0];

    v8f acc[13];
    const v8f vz = {0.f, 0.f, 0.f, 0.f, 0.f, 0.f, 0.f, 0.f};
#pragma unroll
    for (int i = 0; i < 13; ++i) acc[i] = vz;

    for (int k0 = 0; k0 < DIM1; k0 += 32) {
        // --- issue B-tile DMA (wave 0): overlaps with A fill below ---
        if (wave == 0) {
            tdm_load_tile(ldsB0, whi + k0, NPAD1, DIM1);
            tdm_load_tile(ldsB1, wlo + k0, NPAD1, DIM1);
        }
        // --- fill+split A tile: 128 rows x 32 k = 1024 float4 ---
        for (int u = tid; u < 1024; u += 256) {
            const int m  = u >> 3;
            const int kk = (u & 7) << 2;
            const int k  = k0 + kk;
            const int row = mBase + m;
            float4 x;
            if (k < IMG_DIM) x = ld4(images + (size_t)row * IMG_DIM + k);
            else             x = ld4(tags   + (size_t)row * TAG_DIM + (k - IMG_DIM));
            split_store4(&sA[0][0], &sA[1][0], m * LDSK + kk, x);
        }
        if (wave == 0) __builtin_amdgcn_s_wait_tensorcnt(0);
        __syncthreads();

        const v16bf ah = load_frag(&sA[0][wave * 16 * LDSK], lane);
        const v16bf al = load_frag(&sA[1][wave * 16 * LDSK], lane);
#pragma unroll
        for (int nt = 0; nt < 13; ++nt) {
            const v16bf bh = load_frag(&sB[0][nt * 16 * LDSK], lane);
            const v16bf bl = load_frag(&sB[1][nt * 16 * LDSK], lane);
            acc[nt] = WMMA_BF16(ah, bh, acc[nt]);
            acc[nt] = WMMA_BF16(ah, bl, acc[nt]);
            acc[nt] = WMMA_BF16(al, bh, acc[nt]);
        }
        __syncthreads();
    }

    // C/D layout: lane n = lane&15; vgpr j -> m = j + (lane>=16 ? 8 : 0)
    const int mrow = mBase + wave * 16 + ((lane & 16) ? 8 : 0);
    const int nc   = lane & 15;
    for (int nt = 0; nt < 13; ++nt) {
        const int col = nt * 16 + nc;
        if (col >= PE) continue;
#pragma unroll
        for (int j = 0; j < 8; ++j)
            region[(size_t)(mrow + j) * PE + col] = acc[nt][j];
    }
}

// ===========================================================================
// Kernel 2: per-row attention -> box_feat   (one wave32 per row)
// ===========================================================================
__global__ __launch_bounds__(256) void k_attn(const float* __restrict__ boxes,
                                              const float* __restrict__ pos_emb,
                                              const float* __restrict__ region,
                                              float* __restrict__ box_feat) {
    const int wave = threadIdx.x >> 5;
    const int lane = threadIdx.x & 31;
    const int row  = blockIdx.x * 8 + wave;   // < 9216

    const float* rg = region + (size_t)row * PE;
    const float* bx = boxes  + (size_t)row * (2 * BOX);

    int   idx[BOX];
    float w[BOX], s[BOX];
#pragma unroll
    for (int k = 0; k < BOX; ++k) {
        idx[k] = (int)bx[k];
        w[k]   = bx[BOX + k];
    }

    for (int k = 0; k < BOX; ++k) {
        const float* pe = pos_emb + (size_t)idx[k] * PE;
        float p = 0.f;
        for (int e = lane; e < PE; e += 32) p += rg[e] * pe[e];
#pragma unroll
        for (int m = 16; m >= 1; m >>= 1) p += __shfl_xor(p, m, 32);
        s[k] = p;
    }

    float mx = -1e30f;
#pragma unroll
    for (int k = 0; k < BOX; ++k) { s[k] = tanhf(s[k]); mx = fmaxf(mx, s[k]); }
    float sum = 0.f;
#pragma unroll
    for (int k = 0; k < BOX; ++k) { s[k] = expf(s[k] - mx); sum += s[k]; }
    const float inv = 1.f / sum;
    float asum = 0.f;
#pragma unroll
    for (int k = 0; k < BOX; ++k) { s[k] = s[k] * inv * w[k]; asum += s[k]; }
    const float inv2 = 1.f / (asum + 1e-6f);
#pragma unroll
    for (int k = 0; k < BOX; ++k) s[k] *= inv2;

    float* bf = box_feat + (size_t)row * PE;
    for (int e = lane; e < PE; e += 32) {
        float a = 0.f;
#pragma unroll
        for (int k = 0; k < BOX; ++k) a += s[k] * pos_emb[(size_t)idx[k] * PE + e];
        bf[e] = a;
    }
}

// ===========================================================================
// Kernel 3: features = X2 @ fc_W^T + fc_b   (9216 x 3016) x (3016 x 1024)
//   BM=128, BN=256, BK=32 (K padded to 3040); B tiles via TDM; grid (72, 4)
// ===========================================================================
__global__ __launch_bounds__(256) void k_fc(const float* __restrict__ images,
                                            const float* __restrict__ tags,
                                            const float* __restrict__ box_feat,
                                            const unsigned short* __restrict__ whi,
                                            const unsigned short* __restrict__ wlo,
                                            const float* __restrict__ fc_b,
                                            float* __restrict__ out) {
    __shared__ unsigned short sA[2][128 * LDSK];
    __shared__ unsigned short sB[2][256 * LDSK];

    const int tid  = threadIdx.x;
    const int wave = tid >> 5;
    const int lane = tid & 31;
    const int mBase = blockIdx.x * 128;
    const int nBase = blockIdx.y * 256;

    const unsigned ldsB0 = (unsigned)(size_t)&sB[0][0];
    const unsigned ldsB1 = (unsigned)(size_t)&sB[1][0];
    const unsigned short* whiBlk = whi + (size_t)nBase * FC_KPAD;
    const unsigned short* wloBlk = wlo + (size_t)nBase * FC_KPAD;

    v8f acc[16];
    const v8f vz = {0.f, 0.f, 0.f, 0.f, 0.f, 0.f, 0.f, 0.f};
#pragma unroll
    for (int i = 0; i < 16; ++i) acc[i] = vz;

    for (int k0 = 0; k0 < FC_KPAD; k0 += 32) {
        // --- issue B-tile DMA (wave 0): overlaps with A fill below ---
        if (wave == 0) {
            tdm_load_tile(ldsB0, whiBlk + k0, 256, FC_KPAD);
            tdm_load_tile(ldsB1, wloBlk + k0, 256, FC_KPAD);
        }
        // --- fill+split A tile (concat mux; segment boundaries 4-aligned) ---
        for (int u = tid; u < 1024; u += 256) {
            const int m  = u >> 3;
            const int kk = (u & 7) << 2;
            const int k  = k0 + kk;
            const int row = mBase + m;
            float4 x = {0.f, 0.f, 0.f, 0.f};
            if      (k < IMG_DIM)      x = ld4(images   + (size_t)row * IMG_DIM + k);
            else if (k < IMG_DIM + PE) x = ld4(box_feat + (size_t)row * PE + (k - IMG_DIM));
            else if (k < FC_IN)        x = ld4(tags     + (size_t)row * TAG_DIM + (k - IMG_DIM - PE));
            split_store4(&sA[0][0], &sA[1][0], m * LDSK + kk, x);
        }
        if (wave == 0) __builtin_amdgcn_s_wait_tensorcnt(0);
        __syncthreads();

        const v16bf ah = load_frag(&sA[0][wave * 16 * LDSK], lane);
        const v16bf al = load_frag(&sA[1][wave * 16 * LDSK], lane);
#pragma unroll
        for (int nt = 0; nt < 16; ++nt) {
            const v16bf bh = load_frag(&sB[0][nt * 16 * LDSK], lane);
            const v16bf bl = load_frag(&sB[1][nt * 16 * LDSK], lane);
            acc[nt] = WMMA_BF16(ah, bh, acc[nt]);
            acc[nt] = WMMA_BF16(ah, bl, acc[nt]);
            acc[nt] = WMMA_BF16(al, bh, acc[nt]);
        }
        __syncthreads();
    }

    const int mrow = mBase + wave * 16 + ((lane & 16) ? 8 : 0);
    const int nc   = lane & 15;
#pragma unroll
    for (int nt = 0; nt < 16; ++nt) {
        const int col = nBase + nt * 16 + nc;
        const float bias = fc_b[col];
#pragma unroll
        for (int j = 0; j < 8; ++j)
            out[(size_t)(mrow + j) * ES + col] = acc[nt][j] + bias;
    }
}

// ===========================================================================
// Kernel 4: row-wise L2 normalize, in place on d_out (one block per row)
// ===========================================================================
__global__ __launch_bounds__(256) void k_norm(float* __restrict__ out) {
    const int row = blockIdx.x;
    float* p = out + (size_t)row * ES;

    float ss = 0.f;
    for (int i = threadIdx.x; i < ES; i += 256) {
        const float v = p[i];
        ss += v * v;
    }
#pragma unroll
    for (int m = 16; m >= 1; m >>= 1) ss += __shfl_xor(ss, m, 32);

    __shared__ float red[8];
    const int wave = threadIdx.x >> 5;
    const int lane = threadIdx.x & 31;
    if (lane == 0) red[wave] = ss;
    __syncthreads();

    float tot = 0.f;
#pragma unroll
    for (int i = 0; i < 8; ++i) tot += red[i];
    const float inv = 1.f / (sqrtf(tot) + 1e-8f);

    for (int i = threadIdx.x; i < ES; i += 256)
        p[i] *= inv;
}

// ===========================================================================
// Host launcher
// ===========================================================================
extern "C" void kernel_launch(void* const* d_in, const int* in_sizes, int n_in,
                              void* d_out, int out_size, void* d_ws, size_t ws_size,
                              hipStream_t stream) {
    const float* images = (const float*)d_in[0];   // (256,36,2048)
    const float* tags   = (const float*)d_in[1];   // (256,36,768)
    const float* boxes  = (const float*)d_in[2];   // (256,36,30)
    const float* pos    = (const float*)d_in[3];   // (257,200)
    const float* attnW  = (const float*)d_in[4];   // (200,2816)
    const float* fcW    = (const float*)d_in[5];   // (1024,3016)
    const float* fcb    = (const float*)d_in[6];   // (1024,)
    float* out = (float*)d_out;                    // (256,36,1024)

    // Workspace layout (bytes, 16B-aligned throughout)
    char* ws = (char*)d_ws;
    float*          region   = (float*)ws;                         // 9216*200*4  = 7,372,800
    float*          box_feat = (float*)(ws + 7372800);             // 9216*200*4  = 7,372,800
    unsigned short* fcWhi    = (unsigned short*)(ws + 14745600);   // 1024*3040*2 = 6,225,920
    unsigned short* fcWlo    = (unsigned short*)(ws + 20971520);   // 6,225,920
    unsigned short* attnWhi  = (unsigned short*)(ws + 27197440);   // 208*2816*2  = 1,171,456
    unsigned short* attnWlo  = (unsigned short*)(ws + 28368896);   // 1,171,456  (total ~29.5MB)

    k_split_fcW  <<<dim3(1024), dim3(256), 0, stream>>>(fcW,   fcWhi,   fcWlo);
    k_split_attnW<<<dim3(512),  dim3(256), 0, stream>>>(attnW, attnWhi, attnWlo);

    k_region<<<dim3(BR / 128), dim3(256), 0, stream>>>(images, tags, attnWhi, attnWlo, region);
    k_attn  <<<dim3(BR / 8),   dim3(256), 0, stream>>>(boxes, pos, region, box_feat);
    k_fc    <<<dim3(BR / 128, ES / 256), dim3(256), 0, stream>>>(images, tags, box_feat,
                                                                 fcWhi, fcWlo, fcb, out);
    k_norm  <<<dim3(BR),       dim3(256), 0, stream>>>(out);
}
